// RGNN_classifier_79826262164186
// MI455X (gfx1250) — matrix operand order repaired and checked by reading, not compiled, not run on hardware
//
#include <hip/hip_runtime.h>
#include <hip/hip_bf16.h>

// ---------------------------------------------------------------------------
// RGCN (3 layers, R=8, d=256, N=50000, E=800000) + graph-LN + pool + MLP head
// Strategy: aggregate-then-project. Per layer:
//   1) scatter-mean x[src] into agg[N,R,256] (f32 atomics; x is L2-resident)
//   2) pack A = [mean | x] -> bf16 in WMMA-fragment-native 16x32 tiles
//   3) pack B = [W_rel ; W_root] (2304x256) -> bf16 fragment tiles
//   4) WMMA bf16 GEMM [N x 2304]x[2304 x 256], fused bias+residual+ReLU
//      + LayerNorm sum/sumsq reduction in the epilogue
//   5) finalize stats, apply graph-LayerNorm in place
// Then node->graph mean pool (atomics) and a small 3-layer MLP head.
// ---------------------------------------------------------------------------

typedef __attribute__((ext_vector_type(16))) __bf16 v16bf;
typedef __attribute__((ext_vector_type(8)))  float  v8f;

#define N_NODES 50000
#define DIM     256
#define NE      800000
#define NR      8
#define NL      3
#define KDIM    2304          // NR*DIM + DIM
#define MPAD    50048         // N rounded up to 128
#define KT      (KDIM / 32)   // 72 k-tiles
#define NGRAPH  64
#define LN_EPS  1e-5f

__device__ __forceinline__ unsigned short f2bf(float f) {
    unsigned u = __builtin_bit_cast(unsigned, f);
    u += 0x7FFFu + ((u >> 16) & 1u);          // round-to-nearest-even
    return (unsigned short)(u >> 16);
}

// ------------------------------ utility ------------------------------------
__global__ void zero_f32(float* __restrict__ p, size_t n) {
    size_t i = (size_t)blockIdx.x * blockDim.x + threadIdx.x;
    if (i < n) p[i] = 0.f;
}

// ------------------- edge scatter: mean accumulation -----------------------
// one block per edge; 256 lanes carry the 256-wide feature row
__global__ __launch_bounds__(256) void scatter_mean_accum(
    const float* __restrict__ x, const int* __restrict__ src,
    const int* __restrict__ dst, const int* __restrict__ etype,
    float* __restrict__ agg, float* __restrict__ cnt) {
    int e = blockIdx.x;
    int t = threadIdx.x;
    int s = src[e], d = dst[e], r = etype[e];
    size_t ob = ((size_t)d * NR + r) * DIM;
    atomicAdd(&agg[ob + t], x[(size_t)s * DIM + t]);
    if (t == 0) atomicAdd(&cnt[(size_t)d * NR + r], 1.0f);
}

// -------- pack B = [W_rel (2048x256) ; W_root (256x256)] as bf16 frags -----
// tile layout: tile (kb,nb) holds 32(K)x16(N); element (lane,j) = B[kb*32 + j
// + (j&8) + ((lane&16)>>1)][nb*16 + (lane&15)]  (ISA 7.12.2 bf16 B-fragment map)
__global__ __launch_bounds__(256) void pack_B(
    const float* __restrict__ Wrel, const float* __restrict__ Wroot,
    unsigned short* __restrict__ Bpk) {
    size_t idx = (size_t)blockIdx.x * 256 + threadIdx.x;  // KDIM*DIM total
    size_t tile = idx >> 9;
    int pos = (int)(idx & 511);
    int l = pos >> 4, j = pos & 15;
    int kb = (int)(tile >> 4);
    int nb = (int)(tile & 15);
    int k   = kb * 32 + j + (j & 8) + ((l & 16) >> 1);
    int col = nb * 16 + (l & 15);
    float v;
    if (k < NR * DIM) v = Wrel[(size_t)k * DIM + col];
    else              v = Wroot[(size_t)(k - NR * DIM) * DIM + col];
    Bpk[idx] = f2bf(v);
}

// ---- pack A = [mean(agg/cnt) | x] (MPAD x 2304) as bf16 A-fragments -------
// tile (mb,kb): element (lane,j) = A[mb*16 + (lane&15)]
//                                  [kb*32 + j + (j&8) + ((lane&16)>>1)]
__global__ __launch_bounds__(256) void pack_A(
    const float* __restrict__ agg, const float* __restrict__ cnt,
    const float* __restrict__ x, unsigned short* __restrict__ Apk) {
    size_t idx = (size_t)blockIdx.x * 256 + threadIdx.x;
    if (idx >= (size_t)MPAD * KDIM) return;
    size_t tile = idx >> 9;
    int pos = (int)(idx & 511);
    int l = pos >> 4, j = pos & 15;
    int mb = (int)(tile / KT);
    int kb = (int)(tile % KT);
    int row = mb * 16 + (l & 15);
    int k = kb * 32 + j + (j & 8) + ((l & 16) >> 1);
    float v = 0.f;
    if (row < N_NODES) {
        if (k < NR * DIM) {
            int r = k >> 8, c = k & 255;
            float cc = cnt[(size_t)row * NR + r];
            v = agg[((size_t)row * NR + r) * DIM + c] / fmaxf(cc, 1.0f);
        } else {
            v = x[(size_t)row * DIM + (k - NR * DIM)];
        }
    }
    Apk[idx] = f2bf(v);
}

// ------------------- WMMA bf16 GEMM + fused epilogue -----------------------
// block = 8 waves; block tile 128(M) x 64(N); wave tile 32x32 (2x2 frags)
__global__ __launch_bounds__(256) void rgcn_gemm_wmma(
    const unsigned short* __restrict__ Apk, const unsigned short* __restrict__ Bpk,
    const float* __restrict__ xprev, const float* __restrict__ bias,
    float* __restrict__ xout, float* __restrict__ stats, int do_stats) {
    int lane = threadIdx.x & 31;
    int wave = threadIdx.x >> 5;
    int wr = wave & 3;   // row band 0..3
    int wc = wave >> 2;  // col band 0..1
    int rowBase = blockIdx.y * 128 + wr * 32;
    int colBase = blockIdx.x * 64  + wc * 32;
    int mb0 = rowBase >> 4;
    int nb0 = colBase >> 4;

    v8f acc00 = {}, acc01 = {}, acc10 = {}, acc11 = {};
    const v16bf* Afr = (const v16bf*)Apk;   // 32B-aligned fragment chunks
    const v16bf* Bfr = (const v16bf*)Bpk;

    for (int kb = 0; kb < KT; ++kb) {
        v16bf a0 = Afr[((size_t)(mb0 + 0) * KT + kb) * 32 + lane];
        v16bf a1 = Afr[((size_t)(mb0 + 1) * KT + kb) * 32 + lane];
        v16bf b0 = Bfr[((size_t)kb * 16 + (nb0 + 0)) * 32 + lane];
        v16bf b1 = Bfr[((size_t)kb * 16 + (nb0 + 1)) * 32 + lane];
        acc00 = __builtin_amdgcn_wmma_f32_16x16x32_bf16(false, a0, false, b0, (short)0, acc00, false, false);
        acc01 = __builtin_amdgcn_wmma_f32_16x16x32_bf16(false, a0, false, b1, (short)0, acc01, false, false);
        acc10 = __builtin_amdgcn_wmma_f32_16x16x32_bf16(false, a1, false, b0, (short)0, acc10, false, false);
        acc11 = __builtin_amdgcn_wmma_f32_16x16x32_bf16(false, a1, false, b1, (short)0, acc11, false, false);
    }

    // epilogue: out = relu(acc + residual + bias); accumulate LN sum/sumsq
    float s = 0.f, ss = 0.f;
    int rOff = (lane & 16) ? 8 : 0;
    int c0   = lane & 15;
    v8f accs[4] = {acc00, acc01, acc10, acc11};
#pragma unroll
    for (int t = 0; t < 4; ++t) {
        int ti = t >> 1, tj = t & 1;
        int r0 = rowBase + ti * 16 + rOff;
        int c  = colBase + tj * 16 + c0;
#pragma unroll
        for (int v = 0; v < 8; ++v) {
            int row = r0 + v;
            if (row < N_NODES) {
                size_t o = (size_t)row * DIM + c;
                float y = accs[t][v] + xprev[o] + bias[c];
                y = fmaxf(y, 0.f);
                xout[o] = y;
                s += y; ss += y * y;
            }
        }
    }

    if (do_stats) {
        __shared__ float rs[256], rss[256];
        rs[threadIdx.x] = s; rss[threadIdx.x] = ss;
        __syncthreads();
        for (int off = 128; off > 0; off >>= 1) {
            if (threadIdx.x < (unsigned)off) {
                rs[threadIdx.x]  += rs[threadIdx.x + off];
                rss[threadIdx.x] += rss[threadIdx.x + off];
            }
            __syncthreads();
        }
        if (threadIdx.x == 0) {
            atomicAdd(&stats[0], rs[0]);
            atomicAdd(&stats[1], rss[0]);
        }
    }
}

// --------------------------- graph LayerNorm -------------------------------
__global__ void ln_finalize(float* __restrict__ stats) {
    float n = (float)N_NODES * (float)DIM;
    float mu = stats[0] / n;
    float var = fmaxf(stats[1] / n - mu * mu, 0.f);
    stats[2] = mu;
    stats[3] = 1.0f / (sqrtf(var) + LN_EPS);
}

__global__ __launch_bounds__(256) void ln_apply(
    float* __restrict__ x, const float* __restrict__ w,
    const float* __restrict__ b, const float* __restrict__ stats) {
    size_t i = (size_t)blockIdx.x * 256 + threadIdx.x;
    int c = (int)(i & 255);
    float mu = stats[2], inv = stats[3];
    x[i] = w[c] * ((x[i] - mu) * inv) + b[c];
}

// ------------------------------- pooling -----------------------------------
__global__ __launch_bounds__(256) void pool_nodes(
    const float* __restrict__ x, const int* __restrict__ batch,
    float* __restrict__ gsum, float* __restrict__ gcnt) {
    int n = blockIdx.x, t = threadIdx.x;
    int b = batch[n];
    atomicAdd(&gsum[(size_t)b * DIM + t], x[(size_t)n * DIM + t]);
    if (t == 0) atomicAdd(&gcnt[b], 1.0f);
}

// ------------------------------- MLP head ----------------------------------
__global__ __launch_bounds__(256) void fc256(
    const float* __restrict__ in, const float* __restrict__ gcnt,
    const float* __restrict__ W, const float* __restrict__ bias,
    float* __restrict__ out) {
    __shared__ float row[256];
    int g = blockIdx.x, c = threadIdx.x;
    float scale = gcnt ? (1.0f / fmaxf(gcnt[g], 1.0f)) : 1.0f;
    row[c] = in[(size_t)g * DIM + c] * scale;
    __syncthreads();
    float acc = bias[c];
#pragma unroll 8
    for (int k = 0; k < DIM; ++k) acc += row[k] * W[(size_t)k * DIM + c];
    out[(size_t)g * DIM + c] = fmaxf(acc, 0.f);
}

__global__ __launch_bounds__(256) void fc_head(
    const float* __restrict__ in, const float* __restrict__ W,
    const float* __restrict__ b, float* __restrict__ out) {
    int t = threadIdx.x;
    int g = t >> 2, c = t & 3;
    float acc = b[c];
#pragma unroll 8
    for (int k = 0; k < DIM; ++k) acc += in[(size_t)g * DIM + k] * W[(size_t)k * 4 + c];
    out[(size_t)g * 4 + c] = acc;
}

// ------------------------------- driver ------------------------------------
extern "C" void kernel_launch(void* const* d_in, const int* in_sizes, int n_in,
                              void* d_out, int out_size, void* d_ws, size_t ws_size,
                              hipStream_t stream) {
    const float* x0    = (const float*)d_in[0];
    const int*   ei    = (const int*)d_in[1];    // [2,E]
    const int*   et    = (const int*)d_in[2];
    const int*   batch = (const int*)d_in[3];
    const float* Wrel  = (const float*)d_in[4];  // [L,R,d,d]
    const float* Wroot = (const float*)d_in[5];  // [L,d,d]
    const float* bconv = (const float*)d_in[6];  // [L,d]
    const float* lnw   = (const float*)d_in[7];  // [L-1,d]
    const float* lnb   = (const float*)d_in[8];
    const float* W1 = (const float*)d_in[9];
    const float* b1 = (const float*)d_in[10];
    const float* W2 = (const float*)d_in[11];
    const float* b2 = (const float*)d_in[12];
    const float* W3 = (const float*)d_in[13];
    const float* b3 = (const float*)d_in[14];
    float* out = (float*)d_out;

    char* ws = (char*)d_ws;
    auto carve = [&](size_t bytes) -> char* {
        char* p = ws; ws += (bytes + 255) & ~(size_t)255; return p;
    };
    float*          xA    = (float*)carve((size_t)N_NODES * DIM * 4);
    float*          xB    = (float*)carve((size_t)N_NODES * DIM * 4);
    float*          agg   = (float*)carve((size_t)N_NODES * NR * DIM * 4);
    float*          cnt   = (float*)carve((size_t)N_NODES * NR * 4);
    unsigned short* Apk   = (unsigned short*)carve((size_t)MPAD * KDIM * 2);
    unsigned short* Bpk   = (unsigned short*)carve((size_t)KDIM * DIM * 2);
    float*          stats = (float*)carve(256);
    float*          gsum  = (float*)carve((size_t)NGRAPH * DIM * 4);
    float*          gcnt  = (float*)carve((size_t)NGRAPH * 4);
    float*          h1    = (float*)carve((size_t)NGRAPH * DIM * 4);
    float*          h2    = (float*)carve((size_t)NGRAPH * DIM * 4);

    const float* xin = x0;
    float* bufs[2] = {xA, xB};
    for (int i = 0; i < NL; ++i) {
        float* xout = bufs[i & 1];
        size_t naz = (size_t)N_NODES * NR * DIM;
        zero_f32<<<(unsigned)((naz + 255) / 256), 256, 0, stream>>>(agg, naz);
        size_t ncz = (size_t)N_NODES * NR;
        zero_f32<<<(unsigned)((ncz + 255) / 256), 256, 0, stream>>>(cnt, ncz);
        zero_f32<<<1, 256, 0, stream>>>(stats, 4);

        scatter_mean_accum<<<NE, 256, 0, stream>>>(xin, ei, ei + NE, et, agg, cnt);

        pack_B<<<(KDIM * DIM) / 256, 256, 0, stream>>>(
            Wrel + (size_t)i * NR * DIM * DIM, Wroot + (size_t)i * DIM * DIM, Bpk);
        size_t nA = (size_t)MPAD * KDIM;
        pack_A<<<(unsigned)((nA + 255) / 256), 256, 0, stream>>>(agg, cnt, xin, Apk);

        dim3 gg(DIM / 64, MPAD / 128);
        rgcn_gemm_wmma<<<gg, 256, 0, stream>>>(
            Apk, Bpk, xin, bconv + (size_t)i * DIM, xout, stats, (i < NL - 1) ? 1 : 0);

        if (i < NL - 1) {
            ln_finalize<<<1, 1, 0, stream>>>(stats);
            ln_apply<<<(N_NODES * DIM) / 256, 256, 0, stream>>>(
                xout, lnw + (size_t)i * DIM, lnb + (size_t)i * DIM, stats);
        }
        xin = xout;
    }

    // global mean pool per graph
    zero_f32<<<(NGRAPH * DIM) / 256, 256, 0, stream>>>(gsum, (size_t)NGRAPH * DIM);
    zero_f32<<<1, 64, 0, stream>>>(gcnt, (size_t)NGRAPH);
    pool_nodes<<<N_NODES, 256, 0, stream>>>(xin, batch, gsum, gcnt);

    // MLP head
    fc256<<<NGRAPH, 256, 0, stream>>>(gsum, gcnt, W1, b1, h1);
    fc256<<<NGRAPH, 256, 0, stream>>>(h1, nullptr, W2, b2, h2);
    fc_head<<<1, 256, 0, stream>>>(h2, W3, b3, out);
}
